// Model_84164179133279
// MI455X (gfx1250) — compile-verified, matrix operand order checked by reading, not run on hardware
//
#include <hip/hip_runtime.h>
#include <hip/hip_bf16.h>

typedef __bf16 bf16_t;
typedef __attribute__((ext_vector_type(16))) __bf16 v16bf;
typedef __attribute__((ext_vector_type(8)))  float  v8f;
typedef __attribute__((ext_vector_type(4)))  float  f4v;

struct F8 { f4v a, b; };   // 32B, 16B-aligned carrier for a v16bf fragment

#define B_   16
#define L_   2048
#define DM   512
#define H_   8
#define DK_  64
#define DFF  2048
#define COUT 64
#define TOPK 7
#define M_   (B_ * L_)   // 32768

// ---------------------------------------------------------------------------
// Elementwise helpers
// ---------------------------------------------------------------------------
__global__ void k_f32_to_bf16(const float* __restrict__ in, bf16_t* __restrict__ out, int n) {
  int i = blockIdx.x * blockDim.x + threadIdx.x;
  if (i < n) out[i] = (bf16_t)in[i];
}

// pw: [COUT, DM, 3] -> out [DM, COUT] (one tap), bf16
__global__ void k_convW3(const float* __restrict__ pw, bf16_t* __restrict__ out, int tap) {
  int i = blockIdx.x * blockDim.x + threadIdx.x;
  if (i < DM * COUT) {
    int d = i / COUT, c = i % COUT;
    out[i] = (bf16_t)pw[(c * DM + d) * 3 + tap];
  }
}

// ---------------------------------------------------------------------------
// bf16 WMMA GEMM: C[M,N] = A[M,K] @ B[K,N] (+bias) (+resid) (relu) (accum)
// Optional circular row shift of A within each L_-row batch (for circ conv).
// Tile 128x128x32, 8 wave32 per workgroup, each wave owns 16 rows x 128 cols.
// LDS tiles are stored in WMMA *fragment order*: for each row (A) / column (B)
// the 32 K-values are permuted as [0..7,16..23,8..15,24..31] so a lane's
// 16-element fragment is 32 contiguous bytes -> two ds_load_b128.
//   perm(K) = ((K>>3)&1)*16 + (K&7) + ((K>>4)&1)*8
// Any aligned group of 4 K's is contiguous under perm, so A staging moves
// 8 bytes at a time (global b64 load -> LDS b64 store).
// ---------------------------------------------------------------------------
__global__ __launch_bounds__(256) void k_gemm_bf16(
    const bf16_t* __restrict__ A, const bf16_t* __restrict__ Bw,
    const float* __restrict__ bias, const float* __restrict__ resid,
    float* __restrict__ Cf, bf16_t* __restrict__ Cb,
    int M, int N, int K, int relu, int accum, int shiftFlag, int shift)
{
  __shared__ bf16_t aS[128][40];   // [row][Kperm], padded to 80B rows
  __shared__ bf16_t bS[128][40];   // [col][Kperm]

  int tid  = threadIdx.x;
  int n0   = blockIdx.x * 128;
  int m0   = blockIdx.y * 128;
  int lane = tid & 31, wave = tid >> 5;
  int half = lane >> 4, lm = lane & 15;

  v8f acc[8];
#pragma unroll
  for (int t = 0; t < 8; ++t)
#pragma unroll
    for (int e = 0; e < 8; ++e) acc[t][e] = 0.0f;

  for (int k0 = 0; k0 < K; k0 += 32) {
    // ---- Stage A tile: 128 rows x 8 groups of 4 bf16 (8B each) ----
    for (int i = tid; i < 128 * 8; i += 256) {
      int r = i >> 3, g = i & 7;
      int row = m0 + r;
      if (shiftFlag) {
        int bb = row / L_, ll = row % L_;
        ll = (ll + shift + L_) % L_;
        row = bb * L_ + ll;
      }
      int K0 = g * 4;
      unsigned long long vb =
          *(const unsigned long long*)&A[(long)row * K + k0 + K0];
      int pos = ((K0 >> 3) & 1) * 16 + (K0 & 7) + ((K0 >> 4) & 1) * 8;
      *(unsigned long long*)&aS[r][pos] = vb;
    }
    // ---- Stage B tile transposed: 32 k-rows x 32 col-groups of 4 ----
    for (int i = tid; i < 32 * 32; i += 256) {
      int r = i >> 5, c = (i & 31) * 4;
      int col = n0 + c;
      unsigned long long vb = 0ull;
      if (col < N)
        vb = *(const unsigned long long*)&Bw[(long)(k0 + r) * N + col];
      union { unsigned long long u; bf16_t h[4]; } u; u.u = vb;
      int pos = ((r >> 3) & 1) * 16 + (r & 7) + ((r >> 4) & 1) * 8;
      bS[c + 0][pos] = u.h[0];
      bS[c + 1][pos] = u.h[1];
      bS[c + 2][pos] = u.h[2];
      bS[c + 3][pos] = u.h[3];
    }
    __syncthreads();

    if (k0 + 32 < K)
      __builtin_prefetch(&A[(long)(m0 + (tid >> 1)) * K + k0 + 32], 0, 1);

    // ---- A fragment: 32 contiguous bytes (2x ds_load_b128) ----
    int mRow = wave * 16 + lm;
    const f4v* ap = (const f4v*)&aS[mRow][half * 16];
    F8 fa; fa.a = ap[0]; fa.b = ap[1];
    v16bf af = __builtin_bit_cast(v16bf, fa);

#pragma unroll
    for (int nt = 0; nt < 8; ++nt) {
      int colLoc = nt * 16 + lm;
      const f4v* bp = (const f4v*)&bS[colLoc][half * 16];
      F8 fb; fb.a = bp[0]; fb.b = bp[1];
      v16bf bfg = __builtin_bit_cast(v16bf, fb);
      acc[nt] = __builtin_amdgcn_wmma_f32_16x16x32_bf16(
          false, af, false, bfg, (short)0, acc[nt], false, false);
    }
    __syncthreads();
  }

  // Epilogue: C/D layout — VGPR r: lanes0-15 M=r, lanes16-31 M=r+8, N=lane&15
#pragma unroll
  for (int nt = 0; nt < 8; ++nt) {
    int col = n0 + nt * 16 + lm;
    if (col >= N) continue;
#pragma unroll
    for (int r = 0; r < 8; ++r) {
      int row = m0 + wave * 16 + r + 8 * half;
      float v = acc[nt][r];
      if (bias)  v += bias[col];
      if (relu)  v  = v > 0.0f ? v : 0.0f;
      long idx = (long)row * N + col;
      if (resid) v += resid[idx];
      if (Cb) {
        Cb[idx] = (bf16_t)v;
      } else {
        if (accum) v += Cf[idx];
        Cf[idx] = v;
      }
    }
  }
}

// ---------------------------------------------------------------------------
// In-place radix-2 DIT complex FFT, length 2048, 256 threads, LDS resident.
// ---------------------------------------------------------------------------
__device__ __forceinline__ void fft2048(float* zr, float* zi, int tid) {
  for (int t = tid; t < 2048; t += 256) {
    int r = (int)(__brev((unsigned)t) >> 21);
    if (r > t) {
      float tr = zr[t]; zr[t] = zr[r]; zr[r] = tr;
      float ti = zi[t]; zi[t] = zi[r]; zi[r] = ti;
    }
  }
  __syncthreads();
  for (int m = 2; m <= 2048; m <<= 1) {
    int half = m >> 1;
    float ang0 = -6.283185307179586f / (float)m;
    for (int j = tid; j < 1024; j += 256) {
      int grp = j / half;
      int pos = j - grp * half;
      int i0 = grp * m + pos;
      int i1 = i0 + half;
      float s, c;
      __sincosf(ang0 * (float)pos, &s, &c);
      float xr = zr[i1], xi = zi[i1];
      float tr = c * xr - s * xi;
      float ti = c * xi + s * xr;
      float ur = zr[i0], ui = zi[i0];
      zr[i0] = ur + tr; zi[i0] = ui + ti;
      zr[i1] = ur - tr; zi[i1] = ui - ti;
    }
    __syncthreads();
  }
}

// corr[b,h,tau] = irfft( sum_d Qf * conj(Kf) ).  One WG per (b,h).
// Packs z = q + i*k so one FFT per channel; Q,K recovered by Hermitian split.
__global__ __launch_bounds__(256) void k_fftcorr(
    const float* __restrict__ q, const float* __restrict__ k,
    float* __restrict__ corr)
{
  __shared__ float zr[2048], zi[2048], sr[2048], si[2048];
  int tid = threadIdx.x;
  int bh = blockIdx.x;
  int b = bh / H_, h = bh % H_;
  for (int t = tid; t < 2048; t += 256) { sr[t] = 0.0f; si[t] = 0.0f; }
  for (int d = 0; d < DK_; ++d) {
    int col = h * DK_ + d;
    const float* qp = q + (long)b * L_ * DM + col;
    const float* kp = k + (long)b * L_ * DM + col;
    __syncthreads();
    for (int t = tid; t < 2048; t += 256) {
      zr[t] = qp[(long)t * DM];
      zi[t] = kp[(long)t * DM];
    }
    __syncthreads();
    fft2048(zr, zi, tid);
    for (int f = tid; f < 2048; f += 256) {
      int fm = (2048 - f) & 2047;
      float cr = zr[fm], ci = -zi[fm];           // conj(Z[L-f])
      float Qr = 0.5f * (zr[f] + cr), Qi = 0.5f * (zi[f] + ci);
      float Ar = 0.5f * (zr[f] - cr), Ai = 0.5f * (zi[f] - ci); // = i*K
      float Kr = Ai, Ki = -Ar;
      sr[f] += Qr * Kr + Qi * Ki;                // Q * conj(K)
      si[f] += Qi * Kr - Qr * Ki;
    }
  }
  __syncthreads();
  for (int f = tid; f < 2048; f += 256) si[f] = -si[f];  // conj
  __syncthreads();
  fft2048(sr, si, tid);                                   // iFFT via conj trick
  float invL = 1.0f / 2048.0f;
  for (int t = tid; t < 2048; t += 256)
    corr[(long)bh * L_ + t] = sr[t] * invL;
}

// mean over (b,h), then top-7 + softmax (single block).
__global__ __launch_bounds__(256) void k_meantopk(
    const float* __restrict__ corr, float* __restrict__ wout, int* __restrict__ dout)
{
  __shared__ float mean[2048];
  int tid = threadIdx.x;
  for (int t = tid; t < 2048; t += 256) {
    float s = 0.0f;
    for (int bh = 0; bh < B_ * H_; ++bh) s += corr[(long)bh * L_ + t];
    mean[t] = s * (1.0f / (B_ * H_));
  }
  __syncthreads();
  if (tid == 0) {
    int   dsel[TOPK];
    float vsel[TOPK];
    for (int i = 0; i < TOPK; ++i) {
      float best = -3.4e38f; int bd = 0;
      for (int t = 0; t < 2048; ++t) {
        bool used = false;
        for (int j = 0; j < i; ++j) used |= (dsel[j] == t);
        if (!used && mean[t] > best) { best = mean[t]; bd = t; }
      }
      dsel[i] = bd; vsel[i] = best;
    }
    float mx = vsel[0], se = 0.0f, e[TOPK];
    for (int i = 0; i < TOPK; ++i) { e[i] = __expf(vsel[i] - mx); se += e[i]; }
    for (int i = 0; i < TOPK; ++i) { wout[i] = e[i] / se; dout[i] = dsel[i]; }
  }
}

// agg[b,l,c] = sum_j w[j] * v[b,(l+delay_j)%L,c]  -> bf16 (A operand of Wo GEMM)
__global__ void k_aggregate(const float* __restrict__ v, const float* __restrict__ w,
                            const int* __restrict__ del, bf16_t* __restrict__ agg, int n)
{
  int i = blockIdx.x * blockDim.x + threadIdx.x;
  if (i >= n) return;
  int c = i % DM;
  int l = (i / DM) % L_;
  int b = i / (DM * L_);
  float s = 0.0f;
#pragma unroll
  for (int j = 0; j < TOPK; ++j) {
    int ls = l + del[j]; if (ls >= L_) ls -= L_;
    s += w[j] * v[((long)b * L_ + ls) * DM + c];
  }
  agg[i] = (bf16_t)s;
}

// series_decomp, k=25, edge padding; one thread per (b,d), sliding window.
__global__ void k_decomp(const float* __restrict__ y, float* __restrict__ seas,
                         float* __restrict__ trend)
{
  int g = blockIdx.x * blockDim.x + threadIdx.x;
  if (g >= B_ * DM) return;
  int b = g / DM, d = g % DM;
  const float* yp = y + (long)b * L_ * DM + d;
  float W = 0.0f;
  for (int j = -12; j <= 12; ++j) {
    int t = j < 0 ? 0 : j;
    W += yp[(long)t * DM];
  }
  const float inv = 1.0f / 25.0f;
  for (int t = 0; t < L_; ++t) {
    float tr = W * inv;
    float yv = yp[(long)t * DM];
    long  o  = ((long)b * L_ + t) * DM + d;
    seas[o]  = yv - tr;
    trend[o] = tr;
    int tadd = t + 13; if (tadd > L_ - 1) tadd = L_ - 1;
    int tsub = t - 12; if (tsub < 0) tsub = 0;
    W += yp[(long)tadd * DM] - yp[(long)tsub * DM];
  }
}

// ---------------------------------------------------------------------------
extern "C" void kernel_launch(void* const* d_in, const int* in_sizes, int n_in,
                              void* d_out, int out_size, void* d_ws, size_t ws_size,
                              hipStream_t stream) {
  const float* x       = (const float*)d_in[0];
  const float* cross   = (const float*)d_in[1];
  const float* sa_Wq   = (const float*)d_in[2];
  const float* sa_bq   = (const float*)d_in[3];
  const float* sa_Wk   = (const float*)d_in[4];
  const float* sa_bk   = (const float*)d_in[5];
  const float* sa_Wv   = (const float*)d_in[6];
  const float* sa_bv   = (const float*)d_in[7];
  const float* sa_Wo   = (const float*)d_in[8];
  const float* sa_bo   = (const float*)d_in[9];
  const float* ca_Wq   = (const float*)d_in[10];
  const float* ca_bq   = (const float*)d_in[11];
  const float* ca_Wk   = (const float*)d_in[12];
  const float* ca_bk   = (const float*)d_in[13];
  const float* ca_Wv   = (const float*)d_in[14];
  const float* ca_bv   = (const float*)d_in[15];
  const float* ca_Wo   = (const float*)d_in[16];
  const float* ca_bo   = (const float*)d_in[17];
  const float* conv1_W = (const float*)d_in[18];
  const float* conv1_b = (const float*)d_in[19];
  const float* conv2_W = (const float*)d_in[20];
  const float* conv2_b = (const float*)d_in[21];
  const float* pW[3]   = { (const float*)d_in[22], (const float*)d_in[23], (const float*)d_in[24] };

  const size_t NE = (size_t)B_ * L_ * DM;   // 16,777,216

  // Workspace layout
  float* ws_f  = (float*)d_ws;
  float* qf    = ws_f + 0 * NE;
  float* kf    = ws_f + 1 * NE;     // also reused as y = resid + proj buffer
  float* vf    = ws_f + 2 * NE;
  float* x1    = ws_f + 3 * NE;
  float* x2    = ws_f + 4 * NE;
  float* t1    = ws_f + 5 * NE;
  float* t2    = ws_f + 6 * NE;
  float* t3    = ws_f + 7 * NE;
  float* corr  = ws_f + 8 * NE;                       // B*H*L
  float* wvals = corr + (size_t)B_ * H_ * L_;         // 8 floats
  int*   dels  = (int*)(wvals + 8);                   // 8 ints
  bf16_t* xb   = (bf16_t*)(dels + 8);                 // NE bf16
  bf16_t* aggb = xb + NE;                             // NE bf16
  bf16_t* hid  = aggb + NE;                           // M_*DFF bf16
  bf16_t* wtmp = hid + (size_t)M_ * DFF;              // up to DFF*DM bf16

  auto tobf = [&](const float* in, bf16_t* out, int n) {
    k_f32_to_bf16<<<(n + 255) / 256, 256, 0, stream>>>(in, out, n);
  };
  auto gemm = [&](const bf16_t* A, const bf16_t* Bw, const float* bias,
                  const float* resid, float* Cf, bf16_t* Cb,
                  int N, int K, int relu, int accum, int shiftFlag, int shift) {
    dim3 grid((N + 127) / 128, M_ / 128);
    k_gemm_bf16<<<grid, 256, 0, stream>>>(A, Bw, bias, resid, Cf, Cb,
                                          M_, N, K, relu, accum, shiftFlag, shift);
  };
  const int nE = (int)NE;

  // ---------------- Self auto-correlation ----------------
  tobf(x, xb, nE);
  tobf(sa_Wq, wtmp, DM * DM); gemm(xb, wtmp, sa_bq, nullptr, qf, nullptr, DM, DM, 0, 0, 0, 0);
  tobf(sa_Wk, wtmp, DM * DM); gemm(xb, wtmp, sa_bk, nullptr, kf, nullptr, DM, DM, 0, 0, 0, 0);
  tobf(sa_Wv, wtmp, DM * DM); gemm(xb, wtmp, sa_bv, nullptr, vf, nullptr, DM, DM, 0, 0, 0, 0);
  k_fftcorr<<<B_ * H_, 256, 0, stream>>>(qf, kf, corr);
  k_meantopk<<<1, 256, 0, stream>>>(corr, wvals, dels);
  k_aggregate<<<(nE + 255) / 256, 256, 0, stream>>>(vf, wvals, dels, aggb, nE);
  tobf(sa_Wo, wtmp, DM * DM);
  gemm(aggb, wtmp, sa_bo, /*resid=*/x, kf, nullptr, DM, DM, 0, 0, 0, 0);   // kf = x + new_x
  k_decomp<<<(B_ * DM + 255) / 256, 256, 0, stream>>>(kf, x1, t1);

  // ---------------- Cross auto-correlation ----------------
  tobf(x1, xb, nE);
  tobf(ca_Wq, wtmp, DM * DM); gemm(xb, wtmp, ca_bq, nullptr, qf, nullptr, DM, DM, 0, 0, 0, 0);
  tobf(cross, aggb, nE);
  tobf(ca_Wk, wtmp, DM * DM); gemm(aggb, wtmp, ca_bk, nullptr, kf, nullptr, DM, DM, 0, 0, 0, 0);
  tobf(ca_Wv, wtmp, DM * DM); gemm(aggb, wtmp, ca_bv, nullptr, vf, nullptr, DM, DM, 0, 0, 0, 0);
  k_fftcorr<<<B_ * H_, 256, 0, stream>>>(qf, kf, corr);
  k_meantopk<<<1, 256, 0, stream>>>(corr, wvals, dels);
  k_aggregate<<<(nE + 255) / 256, 256, 0, stream>>>(vf, wvals, dels, aggb, nE);
  tobf(ca_Wo, wtmp, DM * DM);
  gemm(aggb, wtmp, ca_bo, /*resid=*/x1, kf, nullptr, DM, DM, 0, 0, 0, 0);  // kf = x1 + new_x
  k_decomp<<<(B_ * DM + 255) / 256, 256, 0, stream>>>(kf, x2, t2);

  // ---------------- FFN ----------------
  tobf(x2, xb, nE);
  tobf(conv1_W, wtmp, DM * DFF);
  gemm(xb, wtmp, conv1_b, nullptr, nullptr, hid, DFF, DM, /*relu=*/1, 0, 0, 0);  // hid bf16
  tobf(conv2_W, wtmp, DFF * DM);
  gemm(hid, wtmp, conv2_b, /*resid=*/x2, kf, nullptr, DM, DFF, 0, 0, 0, 0);      // kf = x2 + y
  k_decomp<<<(B_ * DM + 255) / 256, 256, 0, stream>>>(kf, (float*)d_out, t3);    // x3 -> out

  // ---------------- Trend projections (9 shifted GEMMs, N=64) ----------------
  tobf(t1, xb, nE);
  tobf(t2, aggb, nE);
  tobf(t3, hid, nE);
  bf16_t* tb[3] = { xb, aggb, hid };
  float* out2 = (float*)d_out + NE;
  int first = 1;
  for (int p = 0; p < 3; ++p) {
    for (int tap = 0; tap < 3; ++tap) {
      k_convW3<<<(DM * COUT + 255) / 256, 256, 0, stream>>>(pW[p], wtmp, tap);
      gemm(tb[p], wtmp, nullptr, nullptr, out2, nullptr, COUT, DM,
           0, first ? 0 : 1, /*shiftFlag=*/1, /*shift=*/tap - 1);
      first = 0;
    }
  }
}
